// MultiHeadedAttention_6433861009894
// MI455X (gfx1250) — matrix operand order repaired and checked by reading, not compiled, not run on hardware
//
#include <hip/hip_runtime.h>
#include <hip/hip_bf16.h>

// Fused MHA forward for gfx1250 (MI455X).
// B=2, S=2048, D_MODEL=512, H=8, DHEAD=64. Outputs: out (B,S,512) then attn (B,H,S,S).
// Pipeline: widened WMMA projection GEMMs (f32->f16, Q pre-scaled by 1/sqrt(d)),
// one fused kernel per 16-query block: QK^T (regs) -> masked softmax (shfl+LDS stats)
// -> probs parked in LDS f16 -> vectorized coalesced f32 attn write -> AV from LDS,
// then output GEMM.

typedef __attribute__((ext_vector_type(16))) _Float16 v16h;
typedef __attribute__((ext_vector_type(8)))  _Float16 v8h;
typedef __attribute__((ext_vector_type(8)))  float    v8f;

#define DMODEL 512
#define NHEAD  8
#define DHEAD  64
#define SEQ    2048
#define BATCH  2
#define LDSH   (SEQ + 8)   // padded halfs per LDS prob row (4112B -> +4 banks/row)

// ---- WMMA wrapper -----------------------------------------------------------
static __device__ __forceinline__ v8f wmma_f16(v16h a, v16h b, v8f c) {
    return __builtin_amdgcn_wmma_f32_16x16x32_f16(false, a, false, b, (short)0, c, false, false);
}

// ---- Fragment loaders -------------------------------------------------------
// A 16x32 f16: lanes 0-15 hold K = kb..kb+7 & kb+16..kb+23; lanes 16-31 +8.
static __device__ __forceinline__ v16h load_a_f32(const float* __restrict__ p,
                                                  int row, int ld, int kb, int lane) {
    const float* base = p + (size_t)row * ld + kb + ((lane >> 4) << 3);
    float t[16];
    *(float4*)(t + 0)  = *(const float4*)(base + 0);
    *(float4*)(t + 4)  = *(const float4*)(base + 4);
    *(float4*)(t + 8)  = *(const float4*)(base + 16);
    *(float4*)(t + 12) = *(const float4*)(base + 20);
    v16h f;
#pragma unroll
    for (int i = 0; i < 16; ++i) f[i] = (_Float16)t[i];
    return f;
}

static __device__ __forceinline__ v16h load_a_f16(const _Float16* __restrict__ p,
                                                  int row, int ld, int kb, int lane) {
    const _Float16* base = p + (size_t)row * ld + kb + ((lane >> 4) << 3);
    v8h lo = *(const v8h*)(base);
    v8h hi = *(const v8h*)(base + 16);
    v16h f;
#pragma unroll
    for (int i = 0; i < 8; ++i) { f[i] = lo[i]; f[8 + i] = hi[i]; }
    return f;
}

// B 32x16 f16: lane(&15) = output column; lanes 0-15 hold K = kb..kb+7 & kb+16..kb+23; lanes 16-31 +8.
static __device__ __forceinline__ v16h load_b_f32(const float* __restrict__ p,
                                                  int colrow, int ld, int kb, int lane) {
    const float* base = p + (size_t)colrow * ld + kb + ((lane >> 4) << 3);
    float t[16];
    *(float4*)(t + 0)  = *(const float4*)(base + 0);
    *(float4*)(t + 4)  = *(const float4*)(base + 4);
    *(float4*)(t + 8)  = *(const float4*)(base + 16);
    *(float4*)(t + 12) = *(const float4*)(base + 20);
    v16h f;
#pragma unroll
    for (int i = 0; i < 16; ++i) f[i] = (_Float16)t[i];
    return f;
}

static __device__ __forceinline__ v16h load_b_f16(const _Float16* __restrict__ p,
                                                  int colrow, int ld, int kb, int lane) {
    const _Float16* base = p + (size_t)colrow * ld + kb + ((lane >> 4) << 3);
    v8h lo = *(const v8h*)(base);
    v8h hi = *(const v8h*)(base + 16);
    v16h f;
#pragma unroll
    for (int i = 0; i < 8; ++i) { f[i] = lo[i]; f[8 + i] = hi[i]; }
    return f;
}

// ---- Kernel 1: projection C(4096x512) = (X * W^T + bias) * scale, f16 out ---
// Each wave: 16 rows x 64 cols (4 accs), A fragment reused 4x.
// VTRANS==0: store [b][h][s][d] (Q,K); VTRANS==1: store [b][h][d][s] (V^T).
template <int VTRANS>
__global__ __launch_bounds__(128) void proj_kernel(const float* __restrict__ X,
                                                   const float* __restrict__ W,
                                                   const float* __restrict__ bias,
                                                   _Float16* __restrict__ out,
                                                   float scale) {
    const int lane = threadIdx.x & 31, wave = threadIdx.x >> 5;
    const int l15 = lane & 15, hi = lane >> 4;
    const int tile_n = blockIdx.x * 256 + wave * 64;
    const int tile_m = blockIdx.y * 16;
    v8f acc[4] = {};
    for (int kb = 0; kb < DMODEL; kb += 32) {
        v16h a = load_a_f32(X, tile_m + l15, DMODEL, kb, lane);
#pragma unroll
        for (int j = 0; j < 4; ++j)
            acc[j] = wmma_f16(a, load_b_f32(W, tile_n + j * 16 + l15, DMODEL, kb, lane), acc[j]);
    }
#pragma unroll
    for (int j = 0; j < 4; ++j) {
        const int n = tile_n + j * 16 + l15;
        const float bval = bias[n];
        const int h = n >> 6, d = n & 63;
#pragma unroll
        for (int r = 0; r < 8; ++r) {
            const int m = tile_m + r + (hi << 3);
            const int bidx = m >> 11, s = m & (SEQ - 1);
            const float v = (acc[j][r] + bval) * scale;
            if (VTRANS)
                out[((size_t)((bidx * NHEAD + h) * DHEAD + d)) * SEQ + s] = (_Float16)v;
            else
                out[((size_t)((bidx * NHEAD + h) * SEQ + s)) * DHEAD + d] = (_Float16)v;
        }
    }
}

// ---- Kernel 2: fused QK^T -> masked softmax -> attn write -> AV -------------
// One block = 16 query rows of one (b,h). 8 waves; wave w owns keys [w*256, w*256+256).
__global__ __launch_bounds__(256) void attn_fused_kernel(const _Float16* __restrict__ qf,
                                                         const _Float16* __restrict__ kf,
                                                         const _Float16* __restrict__ vT,
                                                         const int* __restrict__ mask,
                                                         float* __restrict__ attn,
                                                         _Float16* __restrict__ x) {
    __shared__ _Float16 sprob[16 * LDSH];       // 64.25 KB: normalized probs (f16)
    __shared__ float    redbuf[2 * 16 * DHEAD]; // 8 KB: AV partials (2 key halves)
    __shared__ float    wstat[8 * 16];          // per-wave row stats
    __shared__ float    rowstat[16];            // block row max, then 1/sum

    const int tid  = threadIdx.x;
    const int lane = tid & 31, wave = tid >> 5;
    const int l15  = lane & 15, hi = lane >> 4;
    const int qbase = blockIdx.x * 16;
    const int bh    = blockIdx.y;
    const int bidx  = bh >> 3, h = bh & (NHEAD - 1);

    const _Float16* q  = qf + (size_t)bh * SEQ * DHEAD;
    const _Float16* k  = kf + (size_t)bh * SEQ * DHEAD;
    const _Float16* vb = vT + (size_t)bh * DHEAD * SEQ;
    const int*    mrow = mask + bidx * SEQ;

    // Pull this wave's V^T slice toward the WGP while scores/softmax run.
    {
        const int dt = wave & 3, kh = wave >> 2;
        __builtin_prefetch(vb + ((size_t)(dt * 16 + l15)) * SEQ + kh * 1024 + hi * 512, 0, 1);
    }

    // --- Phase 1: scores for 16 q-rows x 256 keys, in registers --------------
    // (Q already carries the 1/sqrt(DHEAD) factor from the projection.)
    const v16h qa0 = load_a_f16(q, qbase + l15, DHEAD, 0, lane);
    const v16h qa1 = load_a_f16(q, qbase + l15, DHEAD, 32, lane);
    const int kbase = wave * 256;
    v8f acc[16];
#pragma unroll
    for (int t = 0; t < 16; ++t) acc[t] = (v8f){};
#pragma unroll
    for (int t = 0; t < 16; ++t) {
        const int krow = kbase + t * 16 + l15;
        acc[t] = wmma_f16(qa0, load_b_f16(k, krow, DHEAD, 0, lane),  acc[t]);
        acc[t] = wmma_f16(qa1, load_b_f16(k, krow, DHEAD, 32, lane), acc[t]);
    }
    // mask (column j is fixed per lane per acc)
#pragma unroll
    for (int t = 0; t < 16; ++t) {
        const int j = kbase + t * 16 + l15;
        if (mrow[j] == 0) {
#pragma unroll
            for (int r = 0; r < 8; ++r) acc[t][r] = -1.0e9f;
        }
    }

    // --- Phase 2: row max (wave-local shfl, then cross-wave via LDS) ---------
    float rmax[8];
#pragma unroll
    for (int r = 0; r < 8; ++r) {
        float m = acc[0][r];
#pragma unroll
        for (int t = 1; t < 16; ++t) m = fmaxf(m, acc[t][r]);
#pragma unroll
        for (int off = 1; off < 16; off <<= 1) m = fmaxf(m, __shfl_xor(m, off, 32));
        rmax[r] = m;
    }
    if (l15 == 0) {
#pragma unroll
        for (int r = 0; r < 8; ++r) wstat[wave * 16 + hi * 8 + r] = rmax[r];
    }
    __syncthreads();
    if (tid < 16) {
        float m = wstat[tid];
        for (int w = 1; w < 8; ++w) m = fmaxf(m, wstat[w * 16 + tid]);
        rowstat[tid] = m;
    }
    __syncthreads();

    // --- Phase 3: exp + row sum ---------------------------------------------
    float mx[8];
#pragma unroll
    for (int r = 0; r < 8; ++r) mx[r] = rowstat[hi * 8 + r];
#pragma unroll
    for (int r = 0; r < 8; ++r) {
        float s = 0.f;
#pragma unroll
        for (int t = 0; t < 16; ++t) { const float p = __expf(acc[t][r] - mx[r]); acc[t][r] = p; s += p; }
#pragma unroll
        for (int off = 1; off < 16; off <<= 1) s += __shfl_xor(s, off, 32);
        if (l15 == 0) wstat[wave * 16 + hi * 8 + r] = s;
    }
    __syncthreads();
    if (tid < 16) {
        float s = 0.f;
        for (int w = 0; w < 8; ++w) s += wstat[w * 16 + tid];
        rowstat[tid] = 1.0f / s;
    }
    __syncthreads();

    // --- Phase 4: normalize, park probs in LDS as f16 ------------------------
    float inv[8];
#pragma unroll
    for (int r = 0; r < 8; ++r) inv[r] = rowstat[hi * 8 + r];
#pragma unroll
    for (int t = 0; t < 16; ++t) {
        const int col = kbase + t * 16 + l15;
#pragma unroll
        for (int r = 0; r < 8; ++r) {
            const int row = hi * 8 + r;
            sprob[row * LDSH + col] = (_Float16)(acc[t][r] * inv[r]);
        }
    }
    __syncthreads();

    // --- Phase 5: vectorized coalesced f32 write of attn ---------------------
    // 16 rows x 2048 cols; each step: 8 halfs from LDS -> 2x float4 to global.
    float* adst = attn + ((size_t)bh * SEQ + qbase) * SEQ;
    for (int g = tid; g < 16 * (SEQ / 8); g += 256) {
        const int row = g >> 8;            // 256 groups of 8 per row
        const int col = (g & 255) << 3;
        const v8h pv = *(const v8h*)&sprob[row * LDSH + col];
        float4 f0, f1;
        f0.x = (float)pv[0]; f0.y = (float)pv[1]; f0.z = (float)pv[2]; f0.w = (float)pv[3];
        f1.x = (float)pv[4]; f1.y = (float)pv[5]; f1.z = (float)pv[6]; f1.w = (float)pv[7];
        float* dp = adst + (size_t)row * SEQ + col;
        *(float4*)(dp + 0) = f0;
        *(float4*)(dp + 4) = f1;
    }

    // --- Phase 6: AV from LDS probs. wave -> (16-col slice, key half) --------
    const int dt = wave & 3;    // output cols dt*16..+15
    const int kh = wave >> 2;   // key half
    v8f oacc = {};
    for (int kb = kh * 1024; kb < kh * 1024 + 1024; kb += 32) {
        const _Float16* abase = &sprob[l15 * LDSH + kb + (hi << 3)];
        const v8h alo = *(const v8h*)(abase);
        const v8h ahi = *(const v8h*)(abase + 16);
        v16h a;
#pragma unroll
        for (int i = 0; i < 8; ++i) { a[i] = alo[i]; a[8 + i] = ahi[i]; }
        oacc = wmma_f16(a, load_b_f16(vb, dt * 16 + l15, SEQ, kb, lane), oacc);
    }
#pragma unroll
    for (int r = 0; r < 8; ++r) {
        const int row = hi * 8 + r;
        redbuf[(kh * 16 + row) * DHEAD + dt * 16 + l15] = oacc[r];
    }
    __syncthreads();

    // --- Phase 7: reduce key halves, write x as f16 (vectorized) -------------
    // 16 rows x 64 cols; each thread: 4 consecutive cols.
    _Float16* xdst = x + ((size_t)(bidx * SEQ + qbase)) * DMODEL + h * DHEAD;
    {
        const int g = tid;                  // 256 threads, 256 groups of 4
        const int row = g >> 4, col = (g & 15) << 2;
        const float4 p0 = *(const float4*)&redbuf[row * DHEAD + col];
        const float4 p1 = *(const float4*)&redbuf[(16 + row) * DHEAD + col];
        _Float16 hv[4];
        hv[0] = (_Float16)(p0.x + p1.x);
        hv[1] = (_Float16)(p0.y + p1.y);
        hv[2] = (_Float16)(p0.z + p1.z);
        hv[3] = (_Float16)(p0.w + p1.w);
        *(uint64_t*)(xdst + (size_t)row * DMODEL + col) = *(const uint64_t*)hv;
    }
}

// ---- Kernel 3: out = x @ Wo^T + bo, f32 out (widened like proj) -------------
__global__ __launch_bounds__(128) void outproj_kernel(const _Float16* __restrict__ X,
                                                      const float* __restrict__ W,
                                                      const float* __restrict__ bias,
                                                      float* __restrict__ out) {
    const int lane = threadIdx.x & 31, wave = threadIdx.x >> 5;
    const int l15 = lane & 15, hi = lane >> 4;
    const int tile_n = blockIdx.x * 256 + wave * 64;
    const int tile_m = blockIdx.y * 16;
    v8f acc[4] = {};
    for (int kb = 0; kb < DMODEL; kb += 32) {
        v16h a = load_a_f16(X, tile_m + l15, DMODEL, kb, lane);
#pragma unroll
        for (int j = 0; j < 4; ++j)
            acc[j] = wmma_f16(a, load_b_f32(W, tile_n + j * 16 + l15, DMODEL, kb, lane), acc[j]);
    }
#pragma unroll
    for (int j = 0; j < 4; ++j) {
        const int n = tile_n + j * 16 + l15;
        const float bval = bias[n];
#pragma unroll
        for (int r = 0; r < 8; ++r) {
            const int m = tile_m + r + (hi << 3);
            out[(size_t)m * DMODEL + n] = acc[j][r] + bval;
        }
    }
}

// ---- Host launcher ----------------------------------------------------------
extern "C" void kernel_launch(void* const* d_in, const int* in_sizes, int n_in,
                              void* d_out, int out_size, void* d_ws, size_t ws_size,
                              hipStream_t stream) {
    const float* query = (const float*)d_in[0];
    const float* key   = (const float*)d_in[1];
    const float* value = (const float*)d_in[2];
    const int*   mask  = (const int*)d_in[3];
    const float* Wq = (const float*)d_in[4];  const float* bq = (const float*)d_in[5];
    const float* Wk = (const float*)d_in[6];  const float* bk = (const float*)d_in[7];
    const float* Wv = (const float*)d_in[8];  const float* bv = (const float*)d_in[9];
    const float* Wo = (const float*)d_in[10]; const float* bo = (const float*)d_in[11];

    float* out  = (float*)d_out;
    float* attn = out + (size_t)BATCH * SEQ * DMODEL;

    const size_t headElems = (size_t)BATCH * NHEAD * SEQ * DHEAD; // 2M f16 each
    _Float16* qf   = (_Float16*)d_ws;
    _Float16* kf   = qf + headElems;
    _Float16* vT   = kf + headElems;
    _Float16* xbuf = vT + headElems;                              // [B][S][512] f16

    dim3 pgrid(DMODEL / 256, (BATCH * SEQ) / 16);                 // (2, 256), 128 thr
    proj_kernel<0><<<pgrid, 128, 0, stream>>>(query, Wq, bq, qf, 0.125f); // 1/sqrt(64)
    proj_kernel<0><<<pgrid, 128, 0, stream>>>(key,   Wk, bk, kf, 1.0f);
    proj_kernel<1><<<pgrid, 128, 0, stream>>>(value, Wv, bv, vT, 1.0f);

    attn_fused_kernel<<<dim3(SEQ / 16, BATCH * NHEAD), 256, 0, stream>>>(
        qf, kf, vT, mask, attn, xbuf);

    outproj_kernel<<<pgrid, 128, 0, stream>>>(xbuf, Wo, bo, out);
}
